// MultiHeadAttention_82626580840938
// MI455X (gfx1250) — matrix operand order repaired and checked by reading, not compiled, not run on hardware
//
#include <hip/hip_runtime.h>
#include <hip/hip_fp16.h>

typedef __attribute__((ext_vector_type(16))) _Float16 v16h;
typedef __attribute__((ext_vector_type(8)))  _Float16 v8h;
typedef __attribute__((ext_vector_type(4)))  _Float16 v4h;
typedef __attribute__((ext_vector_type(8)))  float    v8f;

#define DEV __device__ __forceinline__

constexpr int BATCH = 4;
constexpr int SEQ   = 2048;
constexpr int DM    = 1024;
constexpr int NH    = 16;
constexpr int DH    = 64;
constexpr int MROWS = BATCH * SEQ;   // 8192

DEV v8h ld8h(const _Float16* p) { return *(const v8h*)p; }

DEV v16h cat16(v8h lo, v8h hi) {
  return __builtin_shufflevector(lo, hi, 0,1,2,3,4,5,6,7,8,9,10,11,12,13,14,15);
}

DEV v8f wmma_f16f32(v16h a, v16h b, v8f c) {
  // D = A(16x32 f16) * B(32x16 f16) + C(16x16 f32)
  return __builtin_amdgcn_wmma_f32_16x16x32_f16(false, a, false, b, (short)0, c,
                                                false, false);
}

// ---------------------------------------------------------------------------
// fp32 -> fp16 conversion (vectorized)
// ---------------------------------------------------------------------------
__global__ void cvt_f16_kernel(const float* __restrict__ s,
                               _Float16* __restrict__ d, int n4) {
  int i = blockIdx.x * blockDim.x + threadIdx.x;
  if (i < n4) {
    float4 v = ((const float4*)s)[i];
    v4h o;
    o[0] = (_Float16)v.x; o[1] = (_Float16)v.y;
    o[2] = (_Float16)v.z; o[3] = (_Float16)v.w;
    ((v4h*)d)[i] = o;
  }
}

// ---------------------------------------------------------------------------
// Tiled WMMA GEMM: C[M,N] = A[M,K](f16) * B[K,N](f16) + bias
// Block tile 128x128, 8 waves (2x4), each wave 64x32 (4x2 accumulators).
// MODE 0: write f32 C.
// MODE 1: scatter f16 into Q[b,h,l,dh], K[b,h,l,dh], Vt[b,h,dh,l] layouts.
// ---------------------------------------------------------------------------
template <int MODE>
__global__ __launch_bounds__(256)
void wmma_gemm_kernel(const _Float16* __restrict__ A,
                      const _Float16* __restrict__ B,
                      const float* __restrict__ bias,
                      float* __restrict__ Cout,
                      _Float16* __restrict__ Qh,
                      _Float16* __restrict__ Kh,
                      _Float16* __restrict__ Vt,
                      int Kdim, int Ndim) {
  __shared__ __align__(16) _Float16 lsA[128 * 32];   // [m][k]
  __shared__ __align__(16) _Float16 lsB[128 * 32];   // transposed: [n][k]

  const int tid  = threadIdx.x;
  const int wave = tid >> 5;
  const int lane = tid & 31;
  const int hi   = (lane < 16) ? 0 : 1;
  const int lcol = lane & 15;
  const int n0   = blockIdx.x * 128;
  const int m0   = blockIdx.y * 128;
  const int mw   = (wave >> 2) * 64;
  const int nw   = (wave & 3) * 32;

  const v8f zero = {};
  v8f acc[4][2];
#pragma unroll
  for (int i = 0; i < 4; ++i)
#pragma unroll
    for (int j = 0; j < 2; ++j) acc[i][j] = zero;

  const int arow = tid >> 1;          // 2 threads per A row, 16 halves each
  const int akh  = (tid & 1) * 16;
  const int brow = tid >> 3;          // 8 threads per B row, 16 halves each
  const int bns  = (tid & 7) * 16;

  for (int k0 = 0; k0 < Kdim; k0 += 32) {
    // stage A tile [128x32]
    {
      const _Float16* ga = A + (size_t)(m0 + arow) * Kdim + k0 + akh;
      *(v8h*)&lsA[arow * 32 + akh]     = ld8h(ga);
      *(v8h*)&lsA[arow * 32 + akh + 8] = ld8h(ga + 8);
    }
    // stage B tile [32x128], transposed into LDS as [n][k]
    {
      const _Float16* gb = B + (size_t)(k0 + brow) * Ndim + n0 + bns;
      v8h b0 = ld8h(gb);
      v8h b1 = ld8h(gb + 8);
#pragma unroll
      for (int i = 0; i < 8; ++i) {
        lsB[(bns + i) * 32 + brow]     = b0[i];
        lsB[(bns + 8 + i) * 32 + brow] = b1[i];
      }
      if (k0 + 32 < Kdim)
        __builtin_prefetch(gb + (size_t)32 * Ndim, 0, 3);
    }
    __syncthreads();

    // A fragments: lane row = lcol, K halves {base..base+15}
    v16h af[4];
#pragma unroll
    for (int mi = 0; mi < 4; ++mi) {
      const _Float16* pa = &lsA[(mw + mi * 16 + lcol) * 32 + hi * 16];
      af[mi] = cat16(ld8h(pa), ld8h(pa + 8));
    }
    // B fragments: lane col = lcol, K halves {base..base+15}
    v16h bf[2];
#pragma unroll
    for (int ni = 0; ni < 2; ++ni) {
      const _Float16* pb = &lsB[(nw + ni * 16 + lcol) * 32 + hi * 16];
      bf[ni] = cat16(ld8h(pb), ld8h(pb + 8));
    }
#pragma unroll
    for (int mi = 0; mi < 4; ++mi)
#pragma unroll
      for (int ni = 0; ni < 2; ++ni)
        acc[mi][ni] = wmma_f16f32(af[mi], bf[ni], acc[mi][ni]);
    __syncthreads();
  }

  // epilogue: C element (m,n): row = hi*8 + r, col = lcol
#pragma unroll
  for (int mi = 0; mi < 4; ++mi) {
#pragma unroll
    for (int ni = 0; ni < 2; ++ni) {
      const int ng = n0 + nw + ni * 16 + lcol;
      const float bv = bias[ng];
#pragma unroll
      for (int r = 0; r < 8; ++r) {
        const int mg = m0 + mw + mi * 16 + hi * 8 + r;
        const float v = acc[mi][ni][r] + bv;
        if (MODE == 0) {
          Cout[(size_t)mg * Ndim + ng] = v;
        } else {
          const int sel = ng >> 10;      // 0=Q 1=K 2=V
          const int dm  = ng & 1023;
          const int h   = dm >> 6;
          const int dh  = dm & 63;
          const int b   = mg >> 11;
          const int l   = mg & 2047;
          size_t idx;
          _Float16* dst;
          if (sel == 2) {                // V stored transposed: [b,h,dh,l]
            dst = Vt;
            idx = (((size_t)b * NH + h) * DH + dh) * SEQ + l;
          } else {
            dst = (sel == 0) ? Qh : Kh;  // [b,h,l,dh]
            idx = (((size_t)b * NH + h) * SEQ + l) * DH + dh;
          }
          dst[idx] = (_Float16)v;
        }
      }
    }
  }
}

// ---------------------------------------------------------------------------
// Attention: per (b*h, 16-query tile): scores (WMMA) -> softmax -> attn (f32)
// -> P@V (WMMA, V pre-transposed) -> H (f16, [b, l, h*64+n] layout).
// Dynamic LDS: 16x2048 f32 scores (128 KB) + 8x16x64 f32 partials (32 KB).
// ---------------------------------------------------------------------------
__global__ __launch_bounds__(256)
void attention_kernel(const _Float16* __restrict__ Qh,
                      const _Float16* __restrict__ Kh,
                      const _Float16* __restrict__ Vt,
                      float* __restrict__ attn,
                      _Float16* __restrict__ Hh) {
  extern __shared__ __align__(16) char smem[];
  float* sS = (float*)smem;                        // [16][2048]
  float* sP = (float*)(smem + 16 * SEQ * 4);       // [8][16][64]

  const int tid  = threadIdx.x;
  const int wave = tid >> 5;
  const int lane = tid & 31;
  const int hi   = (lane < 16) ? 0 : 1;
  const int lcol = lane & 15;
  const int q0   = blockIdx.x * 16;
  const int bh   = blockIdx.y;
  const float scale = 0.125f;                      // 1/sqrt(64)

  // ---- phase 1: S = Q K^T * scale ----
  v16h aq[2];
  {
    const _Float16* qp = Qh + ((size_t)bh * SEQ + q0 + lcol) * DH;
#pragma unroll
    for (int s = 0; s < 2; ++s) {
      const _Float16* p = qp + s * 32 + hi * 16;
      aq[s] = cat16(ld8h(p), ld8h(p + 8));
    }
  }
  const v8f zero = {};
  for (int kt = wave; kt < SEQ / 16; kt += 8) {
    const _Float16* kp = Kh + ((size_t)bh * SEQ + kt * 16 + lcol) * DH;
    v8f s8 = zero;
#pragma unroll
    for (int s = 0; s < 2; ++s) {
      const _Float16* p = kp + s * 32 + hi * 16;   // B[k][n] = K[n][k]
      v16h bk = cat16(ld8h(p), ld8h(p + 8));
      s8 = wmma_f16f32(aq[s], bk, s8);
    }
#pragma unroll
    for (int r = 0; r < 8; ++r)
      sS[(hi * 8 + r) * SEQ + kt * 16 + lcol] = s8[r] * scale;
  }
  __syncthreads();

  // ---- phase 2: softmax per row; write fp32 attn ----
#pragma unroll
  for (int rr = 0; rr < 2; ++rr) {
    const int r = wave * 2 + rr;
    float* rowp = sS + (size_t)r * SEQ;
    float mx = -3.0e38f;
    for (int c = lane; c < SEQ; c += 32) mx = fmaxf(mx, rowp[c]);
#pragma unroll
    for (int m = 16; m >= 1; m >>= 1) mx = fmaxf(mx, __shfl_xor(mx, m, 32));
    float sum = 0.0f;
    for (int c = lane; c < SEQ; c += 32) sum += __expf(rowp[c] - mx);
#pragma unroll
    for (int m = 16; m >= 1; m >>= 1) sum += __shfl_xor(sum, m, 32);
    const float inv = 1.0f / sum;
    float* gout = attn + ((size_t)bh * SEQ + q0 + r) * SEQ;
    for (int c = lane; c < SEQ; c += 32) {
      const float p = __expf(rowp[c] - mx) * inv;
      rowp[c] = p;
      gout[c] = p;
    }
  }
  __syncthreads();

  // ---- phase 3: O = P @ V; each wave owns a 256-key slice ----
  v8f c4[4];
#pragma unroll
  for (int i = 0; i < 4; ++i) c4[i] = zero;
  const int kslice = wave * 256;
  for (int ks = 0; ks < 8; ++ks) {
    const int kb = kslice + ks * 32;
    // A fragment: P rows from LDS f32, converted to f16
    v16h ap;
    {
      const float* pr = sS + (size_t)lcol * SEQ + kb + hi * 16;
      const float4 f0 = *(const float4*)(pr);
      const float4 f1 = *(const float4*)(pr + 4);
      const float4 f2 = *(const float4*)(pr + 8);
      const float4 f3 = *(const float4*)(pr + 12);
      ap[0]  = (_Float16)f0.x; ap[1]  = (_Float16)f0.y;
      ap[2]  = (_Float16)f0.z; ap[3]  = (_Float16)f0.w;
      ap[4]  = (_Float16)f1.x; ap[5]  = (_Float16)f1.y;
      ap[6]  = (_Float16)f1.z; ap[7]  = (_Float16)f1.w;
      ap[8]  = (_Float16)f2.x; ap[9]  = (_Float16)f2.y;
      ap[10] = (_Float16)f2.z; ap[11] = (_Float16)f2.w;
      ap[12] = (_Float16)f3.x; ap[13] = (_Float16)f3.y;
      ap[14] = (_Float16)f3.z; ap[15] = (_Float16)f3.w;
    }
    // B fragments: Vt row (fixed d_head, contiguous keys) = 2 x b128 loads
#pragma unroll
    for (int ni = 0; ni < 4; ++ni) {
      const int n = ni * 16 + lcol;
      const _Float16* vp = Vt + ((size_t)bh * DH + n) * SEQ + kb + hi * 16;
      v16h bv = cat16(ld8h(vp), ld8h(vp + 8));
      c4[ni] = wmma_f16f32(ap, bv, c4[ni]);
    }
  }
#pragma unroll
  for (int ni = 0; ni < 4; ++ni)
#pragma unroll
    for (int r = 0; r < 8; ++r)
      sP[wave * 1024 + (hi * 8 + r) * 64 + ni * 16 + lcol] = c4[ni][r];
  __syncthreads();

  // reduce 8 wave partials; emit f16 H in [b, l, h*64+n] layout
  {
    const int e0 = tid * 4;           // 1024 elems, 4 per thread
    const int m  = e0 >> 6;
    const int n  = e0 & 63;
    float s0 = 0, s1 = 0, s2 = 0, s3 = 0;
#pragma unroll
    for (int w = 0; w < 8; ++w) {
      const float* p = sP + w * 1024 + e0;
      s0 += p[0]; s1 += p[1]; s2 += p[2]; s3 += p[3];
    }
    const int b = bh >> 4, h = bh & 15;
    _Float16* dst = Hh + ((size_t)b * SEQ + q0 + m) * DM + h * DH + n;
    v4h o;
    o[0] = (_Float16)s0; o[1] = (_Float16)s1;
    o[2] = (_Float16)s2; o[3] = (_Float16)s3;
    *(v4h*)dst = o;
  }
}

// ---------------------------------------------------------------------------
extern "C" void kernel_launch(void* const* d_in, const int* in_sizes, int n_in,
                              void* d_out, int out_size, void* d_ws,
                              size_t ws_size, hipStream_t stream) {
  const float* x    = (const float*)d_in[0];
  const float* Wqkv = (const float*)d_in[1];
  const float* bqkv = (const float*)d_in[2];
  const float* Wout = (const float*)d_in[3];
  const float* bout = (const float*)d_in[4];

  float* out  = (float*)d_out;
  float* attn = out + (size_t)MROWS * DM;    // second tuple output

  // workspace layout (f16), ~88 MiB total
  _Float16* Xh    = (_Float16*)d_ws;
  _Float16* Wqkvh = Xh    + (size_t)MROWS * DM;
  _Float16* Wouth = Wqkvh + (size_t)DM * 3 * DM;
  _Float16* Qh    = Wouth + (size_t)DM * DM;
  _Float16* Kh    = Qh    + (size_t)MROWS * DM;
  _Float16* Vt    = Kh    + (size_t)MROWS * DM;   // transposed per head
  _Float16* Hh    = Vt    + (size_t)MROWS * DM;

  // stage 1: fp32 -> fp16 conversions
  {
    int n4 = MROWS * DM / 4;
    cvt_f16_kernel<<<(n4 + 255) / 256, 256, 0, stream>>>(x, Xh, n4);
    n4 = DM * 3 * DM / 4;
    cvt_f16_kernel<<<(n4 + 255) / 256, 256, 0, stream>>>(Wqkv, Wqkvh, n4);
    n4 = DM * DM / 4;
    cvt_f16_kernel<<<(n4 + 255) / 256, 256, 0, stream>>>(Wout, Wouth, n4);
  }

  // stage 2: QKV projection (scatter into per-head f16 Q/K/Vt)
  wmma_gemm_kernel<1><<<dim3(3 * DM / 128, MROWS / 128), 256, 0, stream>>>(
      Xh, Wqkvh, bqkv, nullptr, Qh, Kh, Vt, DM, 3 * DM);

  // stage 3: attention (scores + softmax + attn out + P@V)
  attention_kernel<<<dim3(SEQ / 16, BATCH * NH), 256, 160 * 1024, stream>>>(
      Qh, Kh, Vt, attn, Hh);

  // stage 4: output projection
  wmma_gemm_kernel<0><<<dim3(DM / 128, MROWS / 128), 256, 0, stream>>>(
      Hh, Wouth, bout, out, nullptr, nullptr, nullptr, DM, DM);
}
